// ModerateMoE_23398981829024
// MI455X (gfx1250) — compile-verified, matrix operand order checked by reading, not run on hardware
//
#include <hip/hip_runtime.h>
#include <stdint.h>

// ---------------- problem constants ----------------
namespace {
constexpr int kN = 4096;   // B*T tokens
constexpr int kD = 1024;   // model dim
constexpr int kE = 8;      // experts
constexpr int kH = 4096;   // hidden
constexpr int kC = 1280;   // capacity = ceil(1.25*N*K/E)
}

typedef __attribute__((ext_vector_type(16))) __bf16 v16bf;
typedef __attribute__((ext_vector_type(8)))  float  v8f;

union Frag { v16bf v; uint4 q[2]; };

// Load one WMMA 16-bit A/B fragment from LDS laid out [row][K], stride 40 elems.
// Lane holds: first 16B at K = (lane>=16?8:0), second 16B at K = 16 + (lane>=16?8:0).
__device__ __forceinline__ v16bf ld_frag(const __bf16* p) {
  Frag f;
  f.q[0] = *(const uint4*)(p);
  f.q[1] = *(const uint4*)(p + 16);
  return f.v;
}

__device__ __forceinline__ v8f wmma_bf16(v16bf a, v16bf b, v8f c) {
  return __builtin_amdgcn_wmma_f32_16x16x32_bf16(false, a, false, b, (short)0, c, false, false);
}

// pack two fp32 -> two bf16 in one dword (K-pair, low half = first K)
__device__ __forceinline__ uint32_t pack_bf16(float lo, float hi) {
  union { __bf16 h[2]; uint32_t u; } pk;
  pk.h[0] = (__bf16)lo;
  pk.h[1] = (__bf16)hi;
  return pk.u;
}

// ---------------- 1. router: logits, softmax, top-2, gates ----------------
__global__ __launch_bounds__(256) void moe_router(
    const float* __restrict__ x, const float* __restrict__ Wr, const float* __restrict__ br,
    float* __restrict__ logits, float* __restrict__ probs,
    int* __restrict__ idx0, int* __restrict__ idx1,
    float* __restrict__ g0, float* __restrict__ g1)
{
  const int wave = threadIdx.x >> 5;
  const int lane = threadIdx.x & 31;
  const int n = blockIdx.x * 8 + wave;
  const float* xr = x + (size_t)n * kD;
  float acc[8];
#pragma unroll
  for (int e = 0; e < 8; ++e) acc[e] = 0.f;
#pragma unroll 4
  for (int i = 0; i < kD / 32; ++i) {
    const int d = i * 32 + lane;
    const float xv = xr[d];
    const float4 wa = ((const float4*)(Wr + (size_t)d * 8))[0];
    const float4 wb = ((const float4*)(Wr + (size_t)d * 8))[1];
    acc[0] += xv * wa.x; acc[1] += xv * wa.y; acc[2] += xv * wa.z; acc[3] += xv * wa.w;
    acc[4] += xv * wb.x; acc[5] += xv * wb.y; acc[6] += xv * wb.z; acc[7] += xv * wb.w;
  }
#pragma unroll
  for (int e = 0; e < 8; ++e) {
#pragma unroll
    for (int off = 16; off > 0; off >>= 1) acc[e] += __shfl_xor(acc[e], off, 32);
  }
  float lg[8];
#pragma unroll
  for (int e = 0; e < 8; ++e) lg[e] = acc[e] + br[e];
#pragma unroll
  for (int e = 0; e < 8; ++e) if (lane == e) logits[(size_t)n * 8 + e] = lg[e];
  float mx = lg[0];
#pragma unroll
  for (int e = 1; e < 8; ++e) mx = fmaxf(mx, lg[e]);
  float p[8], s = 0.f;
#pragma unroll
  for (int e = 0; e < 8; ++e) { p[e] = __expf(lg[e] - mx); s += p[e]; }
  const float inv = 1.f / s;
#pragma unroll
  for (int e = 0; e < 8; ++e) if (lane == e) probs[(size_t)n * 8 + e] = p[e] * inv;
  int i1 = 0; float v1 = lg[0];
#pragma unroll
  for (int e = 1; e < 8; ++e) if (lg[e] > v1) { v1 = lg[e]; i1 = e; }
  int i2 = -1; float v2 = -3.4e38f;
#pragma unroll
  for (int e = 0; e < 8; ++e) if (e != i1 && lg[e] > v2) { v2 = lg[e]; i2 = e; }
  const float ga = 1.f / (1.f + __expf(v2 - v1));  // softmax([v1,v2])[0]
  if (lane == 0) { idx0[n] = i1; idx1[n] = i2; g0[n] = ga; g1[n] = 1.f - ga; }
}

// ---------------- 2. importance + z_loss (single block, deterministic tree) ----------------
__global__ __launch_bounds__(256) void moe_imp_z(
    const float* __restrict__ probs, const float* __restrict__ logits,
    float* __restrict__ impOut, float* __restrict__ zOut)
{
  __shared__ float simp[256][8];
  __shared__ float sz[256];
  const int t = threadIdx.x;
  float acc[8];
#pragma unroll
  for (int e = 0; e < 8; ++e) acc[e] = 0.f;
  float zs = 0.f;
  for (int r = t; r < kN; r += 256) {
#pragma unroll
    for (int e = 0; e < 8; ++e) {
      acc[e] += probs[(size_t)r * 8 + e];
      const float l = logits[(size_t)r * 8 + e];
      zs += l * l;
    }
  }
#pragma unroll
  for (int e = 0; e < 8; ++e) simp[t][e] = acc[e];
  sz[t] = zs;
  __syncthreads();
  for (int st = 128; st > 0; st >>= 1) {
    if (t < st) {
#pragma unroll
      for (int e = 0; e < 8; ++e) simp[t][e] += simp[t + st][e];
      sz[t] += sz[t + st];
    }
    __syncthreads();
  }
  if (t == 0) {
#pragma unroll
    for (int e = 0; e < 8; ++e) impOut[e] = simp[0][e];
    zOut[0] = sz[0] / (float)(kN * kE) * 1e-4f;
  }
}

// ---------------- 3. deterministic capacity scan (order j = k*N + n) ----------------
__global__ __launch_bounds__(256) void moe_scan(
    const int* __restrict__ idx0, const int* __restrict__ idx1,
    int* __restrict__ posc, int* __restrict__ keep, float* __restrict__ loadOut)
{
  __shared__ int ce[256];
  __shared__ int waveCnt[8][8];
  __shared__ int baseCnt[8];
  const int t = threadIdx.x;
  if (t < 8) baseCnt[t] = 0;
  __syncthreads();
  for (int c = 0; c < (2 * kN) / 256; ++c) {
    const int j = c * 256 + t;
    const int n = j & (kN - 1);
    const int k = j >> 12;
    const int e = k ? idx1[n] : idx0[n];
    ce[t] = e;
    __syncthreads();
    const int w = t >> 5;
    int rank = 0;
    for (int s = w * 32; s < t; ++s) rank += (ce[s] == e) ? 1 : 0;
    if (t < 64) {
      const int w2 = t >> 3, e2 = t & 7;
      int cnt = 0;
      for (int s = 0; s < 32; ++s) cnt += (ce[w2 * 32 + s] == e2) ? 1 : 0;
      waveCnt[w2][e2] = cnt;
    }
    __syncthreads();
    int pos = baseCnt[e] + rank;
    for (int ww = 0; ww < w; ++ww) pos += waveCnt[ww][e];
    keep[j] = (pos < kC) ? 1 : 0;
    posc[j] = (pos < kC - 1) ? pos : (kC - 1);
    __syncthreads();
    if (t < 8) {
      int tot = 0;
      for (int ww = 0; ww < 8; ++ww) tot += waveCnt[ww][t];
      baseCnt[t] += tot;
    }
    __syncthreads();
  }
  if (t < 8) loadOut[t] = (float)((baseCnt[t] < kC) ? baseCnt[t] : kC);
}

// ---------------- 4. per-token combine weights ----------------
__global__ __launch_bounds__(256) void moe_weights(
    const int* __restrict__ keep, const float* __restrict__ g0, const float* __restrict__ g1,
    float* __restrict__ w0, float* __restrict__ w1)
{
  const int n = blockIdx.x * 256 + threadIdx.x;
  const float a0 = keep[n]      ? g0[n] : 0.f;
  const float a1 = keep[kN + n] ? g1[n] : 0.f;
  const float denom = fmaxf(a0 + a1, 1e-8f);
  w0[n] = a0 / denom;
  w1[n] = a1 / denom;
}

// ---------------- 5. fused LayerNorm + dispatch (write bf16 xn into [E][C][D]) ------------
__global__ __launch_bounds__(256) void moe_ln_dispatch(
    const float* __restrict__ x, const int* __restrict__ idx0, const int* __restrict__ idx1,
    const int* __restrict__ keep, const int* __restrict__ posc,
    const float* __restrict__ lnS, const float* __restrict__ lnB, __bf16* __restrict__ xnb)
{
  const int j = blockIdx.x;
  if (!keep[j]) return;
  const int n = j & (kN - 1);
  const int k = j >> 12;
  const int e = k ? idx1[n] : idx0[n];
  const int slot = posc[j];
  const int t = threadIdx.x;
  const float4 v = ((const float4*)(x + (size_t)n * kD))[t];
  __shared__ float rs[256], rss[256];
  rs[t]  = v.x + v.y + v.z + v.w;
  rss[t] = v.x * v.x + v.y * v.y + v.z * v.z + v.w * v.w;
  __syncthreads();
  for (int st = 128; st > 0; st >>= 1) {
    if (t < st) { rs[t] += rs[t + st]; rss[t] += rss[t + st]; }
    __syncthreads();
  }
  const float mu = rs[0] * (1.f / kD);
  const float var = rss[0] * (1.f / kD) - mu * mu;
  const float rstd = rsqrtf(var + 1e-5f);
  const int d = t * 4;
  const float4 sc = *(const float4*)(lnS + (size_t)e * kD + d);
  const float4 bs = *(const float4*)(lnB + (size_t)e * kD + d);
  union { __bf16 h[4]; uint2 u; } o;
  o.h[0] = (__bf16)((v.x - mu) * rstd * sc.x + bs.x);
  o.h[1] = (__bf16)((v.y - mu) * rstd * sc.y + bs.y);
  o.h[2] = (__bf16)((v.z - mu) * rstd * sc.z + bs.z);
  o.h[3] = (__bf16)((v.w - mu) * rstd * sc.w + bs.w);
  *(uint2*)(xnb + ((size_t)e * kC + slot) * kD + d) = o.u;
}

// ---------------- 6. GEMM1 fused with SwiGLU: v = a*sigmoid(g), bf16 out ----------------
// block tile: M=128, N=64 (a-columns [n0,n0+64) and g-columns [H+n0,...)).
// 8 waves = 4(M) x 2(N); wave tile 32x32 per output -> 8 WMMA C-tiles/wave.
// Double-buffered LDS (one barrier per K-step); B transpose uses packed K-pair b32 stores.
__global__ __launch_bounds__(256) void moe_gemm1(
    const __bf16* __restrict__ xnb, const float* __restrict__ W1,
    const float* __restrict__ b1, __bf16* __restrict__ vbuf)
{
  constexpr int BM = 128, BN = 64, LDSS = 40;
  const int bid = blockIdx.x;
  const int e    = bid / ((kC / BM) * (kH / BN));
  const int r    = bid % ((kC / BM) * (kH / BN));
  const int m0   = (r / (kH / BN)) * BM;
  const int n0   = (r % (kH / BN)) * BN;

  __shared__ __bf16 As[2][BM * LDSS];
  __shared__ __bf16 Bsh[2][2][BN * LDSS];

  const int tid = threadIdx.x;
  const int lane = tid & 31, w = tid >> 5;
  const int wm = w >> 1, wn = w & 1;
  const int lrow = lane & 15, lh = lane >> 4;

  const __bf16* Abase = xnb + ((size_t)e * kC + m0) * (size_t)kD;
  const float*  Bbase = W1 + (size_t)e * kD * (2 * kH);

  auto stage = [&](int buf, int k0) {
    {  // A tile [BM][32] bf16, verbatim copy (K-contiguous)
      const int row = tid >> 1, half = tid & 1;
      const uint4* src = (const uint4*)(Abase + (size_t)row * kD + k0 + half * 16);
      uint4* dst = (uint4*)(As[buf] + row * LDSS + half * 16);
      dst[0] = src[0];
      dst[1] = src[1];
    }
    {  // B tiles: 2 K-rows x 4 cols per thread, fp32->bf16 packed along K
      const int kk  = (tid >> 4) * 2;   // even K row
      const int grp = tid & 15;         // 4 columns each
#pragma unroll
      for (int hh = 0; hh < 2; ++hh) {
        const float* s0 = Bbase + (size_t)(k0 + kk) * (2 * kH) + hh * kH + n0 + grp * 4;
        const float4 r0 = *(const float4*)(s0);
        const float4 r1 = *(const float4*)(s0 + 2 * kH);
        __bf16* bp = Bsh[buf][hh];
        const int cb = grp * 4;
        *(uint32_t*)(bp + (cb + 0) * LDSS + kk) = pack_bf16(r0.x, r1.x);
        *(uint32_t*)(bp + (cb + 1) * LDSS + kk) = pack_bf16(r0.y, r1.y);
        *(uint32_t*)(bp + (cb + 2) * LDSS + kk) = pack_bf16(r0.z, r1.z);
        *(uint32_t*)(bp + (cb + 3) * LDSS + kk) = pack_bf16(r0.w, r1.w);
        if (k0 + 64 < kD)  // prefetch tile-after-next weights
          __builtin_prefetch(s0 + (size_t)64 * (2 * kH), 0, 1);
      }
    }
  };

  v8f accA[2][2], accG[2][2];
  const v8f vzero = {0.f,0.f,0.f,0.f,0.f,0.f,0.f,0.f};
#pragma unroll
  for (int a = 0; a < 2; ++a)
#pragma unroll
    for (int b = 0; b < 2; ++b) { accA[a][b] = vzero; accG[a][b] = vzero; }

  stage(0, 0);
  __syncthreads();

  for (int k0 = 0; k0 < kD; k0 += 32) {
    const int buf = (k0 >> 5) & 1;
    if (k0 + 32 < kD) stage(buf ^ 1, k0 + 32);

    v16bf afr[2], bfa[2], bfg[2];
#pragma unroll
    for (int mt = 0; mt < 2; ++mt)
      afr[mt] = ld_frag(As[buf] + (wm * 32 + mt * 16 + lrow) * LDSS + lh * 8);
#pragma unroll
    for (int nt = 0; nt < 2; ++nt) {
      bfa[nt] = ld_frag(Bsh[buf][0] + (wn * 32 + nt * 16 + lrow) * LDSS + lh * 8);
      bfg[nt] = ld_frag(Bsh[buf][1] + (wn * 32 + nt * 16 + lrow) * LDSS + lh * 8);
    }
#pragma unroll
    for (int mt = 0; mt < 2; ++mt)
#pragma unroll
      for (int nt = 0; nt < 2; ++nt) {
        accA[mt][nt] = wmma_bf16(afr[mt], bfa[nt], accA[mt][nt]);
        accG[mt][nt] = wmma_bf16(afr[mt], bfg[nt], accG[mt][nt]);
      }
    __syncthreads();  // next-buffer writes done AND this-buffer reads done
  }

  // epilogue: add b1, SwiGLU, store bf16 v
#pragma unroll
  for (int mt = 0; mt < 2; ++mt) {
#pragma unroll
    for (int nt = 0; nt < 2; ++nt) {
      const int ncol = n0 + wn * 32 + nt * 16 + lrow;
      const float ba = b1[(size_t)e * 2 * kH + ncol];
      const float bg = b1[(size_t)e * 2 * kH + kH + ncol];
      const int mbase = m0 + wm * 32 + mt * 16 + ((lane >= 16) ? 8 : 0);
#pragma unroll
      for (int i = 0; i < 8; ++i) {
        const float a = accA[mt][nt][i] + ba;
        const float g = accG[mt][nt][i] + bg;
        const float vv = a * (1.f / (1.f + __expf(-g)));
        vbuf[((size_t)e * kC + mbase + i) * kH + ncol] = (__bf16)vv;
      }
    }
  }
}

// ---------------- 7. GEMM2: expert_out = v @ W2 + b2 (fp32 out) ----------------
// block tile: M=128, N=128; 8 waves = 4(M) x 2(N); wave tile 32x64 -> 8 C-tiles.
// Same double-buffered / packed-transpose scheme.
__global__ __launch_bounds__(256) void moe_gemm2(
    const __bf16* __restrict__ vbuf, const float* __restrict__ W2,
    const float* __restrict__ b2, float* __restrict__ eout)
{
  constexpr int BM = 128, BN = 128, LDSS = 40;
  const int bid = blockIdx.x;
  const int e    = bid / ((kC / BM) * (kD / BN));
  const int r    = bid % ((kC / BM) * (kD / BN));
  const int m0   = (r / (kD / BN)) * BM;
  const int n0   = (r % (kD / BN)) * BN;

  __shared__ __bf16 As[2][BM * LDSS];
  __shared__ __bf16 Bs[2][BN * LDSS];

  const int tid = threadIdx.x;
  const int lane = tid & 31, w = tid >> 5;
  const int wm = w >> 1, wn = w & 1;
  const int lrow = lane & 15, lh = lane >> 4;

  const __bf16* Abase = vbuf + ((size_t)e * kC + m0) * (size_t)kH;
  const float*  Bbase = W2 + (size_t)e * kH * kD;

  auto stage = [&](int buf, int k0) {
    {
      const int row = tid >> 1, half = tid & 1;
      const uint4* src = (const uint4*)(Abase + (size_t)row * kH + k0 + half * 16);
      uint4* dst = (uint4*)(As[buf] + row * LDSS + half * 16);
      dst[0] = src[0];
      dst[1] = src[1];
    }
    {
      const int kk  = (tid >> 4) * 2;   // even K row
      const int grp = tid & 15;         // 8 columns each
      const float* s0 = Bbase + (size_t)(k0 + kk) * kD + n0 + grp * 8;
      const float* s1 = s0 + kD;
      const float4 a0 = ((const float4*)s0)[0];
      const float4 a1 = ((const float4*)s0)[1];
      const float4 c0 = ((const float4*)s1)[0];
      const float4 c1 = ((const float4*)s1)[1];
      __bf16* bp = Bs[buf];
      const int cb = grp * 8;
      *(uint32_t*)(bp + (cb + 0) * LDSS + kk) = pack_bf16(a0.x, c0.x);
      *(uint32_t*)(bp + (cb + 1) * LDSS + kk) = pack_bf16(a0.y, c0.y);
      *(uint32_t*)(bp + (cb + 2) * LDSS + kk) = pack_bf16(a0.z, c0.z);
      *(uint32_t*)(bp + (cb + 3) * LDSS + kk) = pack_bf16(a0.w, c0.w);
      *(uint32_t*)(bp + (cb + 4) * LDSS + kk) = pack_bf16(a1.x, c1.x);
      *(uint32_t*)(bp + (cb + 5) * LDSS + kk) = pack_bf16(a1.y, c1.y);
      *(uint32_t*)(bp + (cb + 6) * LDSS + kk) = pack_bf16(a1.z, c1.z);
      *(uint32_t*)(bp + (cb + 7) * LDSS + kk) = pack_bf16(a1.w, c1.w);
      if (k0 + 64 < kH)
        __builtin_prefetch(s0 + (size_t)64 * kD, 0, 1);
    }
  };

  v8f acc[2][4];
  const v8f vzero = {0.f,0.f,0.f,0.f,0.f,0.f,0.f,0.f};
#pragma unroll
  for (int a = 0; a < 2; ++a)
#pragma unroll
    for (int b = 0; b < 4; ++b) acc[a][b] = vzero;

  stage(0, 0);
  __syncthreads();

  for (int k0 = 0; k0 < kH; k0 += 32) {
    const int buf = (k0 >> 5) & 1;
    if (k0 + 32 < kH) stage(buf ^ 1, k0 + 32);

    v16bf afr[2], bfr[4];
#pragma unroll
    for (int mt = 0; mt < 2; ++mt)
      afr[mt] = ld_frag(As[buf] + (wm * 32 + mt * 16 + lrow) * LDSS + lh * 8);
#pragma unroll
    for (int nt = 0; nt < 4; ++nt)
      bfr[nt] = ld_frag(Bs[buf] + (wn * 64 + nt * 16 + lrow) * LDSS + lh * 8);
#pragma unroll
    for (int mt = 0; mt < 2; ++mt)
#pragma unroll
      for (int nt = 0; nt < 4; ++nt)
        acc[mt][nt] = wmma_bf16(afr[mt], bfr[nt], acc[mt][nt]);
    __syncthreads();
  }

#pragma unroll
  for (int mt = 0; mt < 2; ++mt) {
#pragma unroll
    for (int nt = 0; nt < 4; ++nt) {
      const int ncol = n0 + wn * 64 + nt * 16 + lrow;
      const float bb = b2[(size_t)e * kD + ncol];
      const int mbase = m0 + wm * 32 + mt * 16 + ((lane >= 16) ? 8 : 0);
#pragma unroll
      for (int i = 0; i < 8; ++i)
        eout[((size_t)e * kC + mbase + i) * kD + ncol] = acc[mt][nt][i] + bb;
    }
  }
}

// ---------------- 8. combine: y[n] = w0*out[e0,s0] + w1*out[e1,s1] ----------------
__global__ __launch_bounds__(256) void moe_combine(
    const float* __restrict__ eout, const int* __restrict__ idx0, const int* __restrict__ idx1,
    const int* __restrict__ posc, const float* __restrict__ w0, const float* __restrict__ w1,
    float* __restrict__ y)
{
  const int n = blockIdx.x;
  const int t = threadIdx.x;
  const int e0 = idx0[n], e1 = idx1[n];
  const int s0 = posc[n], s1 = posc[kN + n];
  const float a0 = w0[n], a1 = w1[n];
  const float4 p = ((const float4*)(eout + ((size_t)e0 * kC + s0) * kD))[t];
  const float4 q = ((const float4*)(eout + ((size_t)e1 * kC + s1) * kD))[t];
  float4 o;
  o.x = a0 * p.x + a1 * q.x;
  o.y = a0 * p.y + a1 * q.y;
  o.z = a0 * p.z + a1 * q.z;
  o.w = a0 * p.w + a1 * q.w;
  ((float4*)(y + (size_t)n * kD))[t] = o;
}

// ---------------- 9. balance loss ----------------
__global__ void moe_balance(const float* __restrict__ imp, const float* __restrict__ loadv,
                            float* __restrict__ out)
{
  if (threadIdx.x == 0 && blockIdx.x == 0) {
    float res[2];
    const float* src[2] = { imp, loadv };
    for (int s = 0; s < 2; ++s) {
      float m = 0.f;
      for (int e = 0; e < 8; ++e) m += src[s][e];
      m *= (1.f / 8.f);
      float var = 0.f;
      for (int e = 0; e < 8; ++e) { const float d = src[s][e] - m; var += d * d; }
      var *= (1.f / 8.f);
      const float mm = fmaxf(m, 1e-9f);
      res[s] = var / (mm * mm + 1e-9f);
    }
    out[0] = 0.5f * (res[0] + res[1]);
  }
}

// ---------------- launch ----------------
extern "C" void kernel_launch(void* const* d_in, const int* in_sizes, int n_in,
                              void* d_out, int out_size, void* d_ws, size_t ws_size,
                              hipStream_t stream) {
  const float* h    = (const float*)d_in[0];
  const float* Wr   = (const float*)d_in[1];
  const float* br   = (const float*)d_in[2];
  const float* lnS  = (const float*)d_in[3];
  const float* lnB  = (const float*)d_in[4];
  const float* W1   = (const float*)d_in[5];
  const float* b1   = (const float*)d_in[6];
  const float* W2   = (const float*)d_in[7];
  const float* b2   = (const float*)d_in[8];
  float* out = (float*)d_out;

  char* ws = (char*)d_ws;
  size_t off = 0;
  auto alloc = [&](size_t bytes) { void* p = ws + off; off = (off + bytes + 255) & ~(size_t)255; return p; };
  float* logits = (float*)alloc((size_t)kN * kE * 4);
  float* probs  = (float*)alloc((size_t)kN * kE * 4);
  int*   idx0   = (int*)  alloc((size_t)kN * 4);
  int*   idx1   = (int*)  alloc((size_t)kN * 4);
  float* g0     = (float*)alloc((size_t)kN * 4);
  float* g1     = (float*)alloc((size_t)kN * 4);
  float* w0     = (float*)alloc((size_t)kN * 4);
  float* w1     = (float*)alloc((size_t)kN * 4);
  int*   posc   = (int*)  alloc((size_t)2 * kN * 4);
  int*   keep   = (int*)  alloc((size_t)2 * kN * 4);
  float* imp    = (float*)alloc(64);
  float* loadv  = (float*)alloc(64);
  const size_t xnBytes = (size_t)kE * kC * kD * 2;
  __bf16* xnb   = (__bf16*)alloc(xnBytes);
  __bf16* vbuf  = (__bf16*)alloc((size_t)kE * kC * kH * 2);
  float*  eout  = (float*) alloc((size_t)kE * kC * kD * 4);

  // zero xn buffer so never-filled capacity slots are finite (their outputs get weight 0)
  hipMemsetAsync(xnb, 0, xnBytes, stream);

  moe_router<<<kN / 8, 256, 0, stream>>>(h, Wr, br, logits, probs, idx0, idx1, g0, g1);
  moe_imp_z<<<1, 256, 0, stream>>>(probs, logits, imp, out + (size_t)kN * kD + 1);
  moe_scan<<<1, 256, 0, stream>>>(idx0, idx1, posc, keep, loadv);
  moe_weights<<<kN / 256, 256, 0, stream>>>(keep, g0, g1, w0, w1);
  moe_ln_dispatch<<<2 * kN, 256, 0, stream>>>(h, idx0, idx1, keep, posc, lnS, lnB, xnb);
  moe_gemm1<<<kE * (kC / 128) * (kH / 64), 256, 0, stream>>>(xnb, W1, b1, vbuf);
  moe_gemm2<<<kE * (kC / 128) * (kD / 128), 256, 0, stream>>>(vbuf, W2, b2, eout);
  moe_combine<<<kN, 256, 0, stream>>>(eout, idx0, idx1, posc, w0, w1, out);
  moe_balance<<<1, 64, 0, stream>>>(imp, loadv, out + (size_t)kN * kD);
}
